// SE3TransformerInteractionBlock_47571057770949
// MI455X (gfx1250) — matrix operand order repaired and checked by reading, not compiled, not run on hardware
//
#include <hip/hip_runtime.h>
#include <math.h>

typedef __attribute__((ext_vector_type(16))) __bf16 v16bf;
typedef __attribute__((ext_vector_type(8)))  float  v8f;

#define INV_SQRT32  0.17677669529663689f
#define INV_SQRT3   0.57735026918962576f
#define INV_FAN     0.14433756729740643f   /* 1/sqrt(48) */
#define INV_FAND    0.11180339887498949f   /* 1/sqrt(80) */
#define INV_SQRT_HD 0.22360679774997896f   /* 1/sqrt(20) */

static __device__ __forceinline__ float siluf(float x) { return x / (1.f + __expf(-x)); }

static __device__ __forceinline__ v8f wmma_bf16(v16bf a, v16bf b, v8f c) {
  return __builtin_amdgcn_wmma_f32_16x16x32_bf16(false, a, false, b, (short)0, c, false,
                                                 false);
}

// ---------------------------------------------------------------------------
// Preprocess: Wk2/Wv2 (64x2304 f32, row-major) -> bf16, WMMA-B swizzled:
//   dst[mat][cg(144)][kc(2)][lane(32)][j(16)],
//   value = W[kc*32 + (lane>>4)*16 + j][cg*16 + (lane&15)]
// ---------------------------------------------------------------------------
__global__ __launch_bounds__(256) void prep_kernel(const float* __restrict__ Wk2,
                                                   const float* __restrict__ Wv2,
                                                   __bf16* __restrict__ dst) {
  int idx = blockIdx.x * 256 + threadIdx.x;
  if (idx >= 2 * 147456) return;
  int mat  = idx / 147456;
  int rem  = idx % 147456;
  int cg   = rem / 1024;
  int r2   = rem % 1024;
  int kc   = r2 / 512;
  int r3   = r2 % 512;
  int lane = r3 / 16;
  int j    = r3 % 16;
  int col  = cg * 16 + (lane & 15);
  int k    = kc * 32 + (lane >> 4) * 16 + j;
  const float* W = mat ? Wv2 : Wk2;
  dst[idx] = (__bf16)W[k * 2304 + col];
}

// ---------------------------------------------------------------------------
// Init accumulators
// ---------------------------------------------------------------------------
__global__ __launch_bounds__(256) void init_kernel(float* __restrict__ segmax,
                                                   float* __restrict__ denom,
                                                   float* __restrict__ agg0,
                                                   float* __restrict__ agg1, int N) {
  int idx = blockIdx.x * 256 + threadIdx.x;
  if (idx < N * 4) { segmax[idx] = -10.f; denom[idx] = 0.f; }
  if (idx < N * 32) agg0[idx] = 0.f;
  if (idx < N * 48) agg1[idx] = 0.f;
}

// ---------------------------------------------------------------------------
// Per-node queries: q0[n,h,w] , q1[n,h,w,i]
// ---------------------------------------------------------------------------
__global__ __launch_bounds__(256) void q_kernel(const float* __restrict__ nf,
                                                const float* __restrict__ Wq0,
                                                const float* __restrict__ Wq1,
                                                float* __restrict__ q0g,
                                                float* __restrict__ q1g, int N) {
  int idx = blockIdx.x * 256 + threadIdx.x;
  if (idx >= N * 80) return;
  int n = idx / 80, c = idx % 80;
  const float* row = nf + (size_t)n * 80;
  if (c < 32) {
    int h = c >> 3, w = c & 7;
    float acc = 0.f;
    #pragma unroll
    for (int u = 0; u < 32; ++u) acc += row[u] * Wq0[(h * 32 + u) * 8 + w];
    q0g[n * 32 + c] = acc * INV_SQRT32;
  } else {
    int t = c - 32, h = t / 12, rem = t % 12, w = rem / 3, ii = rem % 3;
    float acc = 0.f;
    #pragma unroll
    for (int u = 0; u < 16; ++u) acc += row[32 + u * 3 + ii] * Wq1[(h * 16 + u) * 4 + w];
    q1g[n * 48 + t] = acc * 0.25f;
  }
}

// ---------------------------------------------------------------------------
// Per-matrix group sweep over 144 WMMA column-groups, with TWO 16-edge M-tiles
// sharing each B fragment (2x B reuse from L2). u-reduction via ds_add_f32.
// Feature arrays have a 512/256/768-float stride per edge-tile; accumulators
// have 512/256/768-float strides per edge-tile.
// ---------------------------------------------------------------------------
static __device__ __forceinline__ void tp_groups(
    int wave, int lane, int m0b,
    const float* hid,                  // [64][32]  (k, edge-in-block)
    const __bf16* __restrict__ wbase,  // swizzled bf16 weights for this matrix
    const float* __restrict__ b2,      // second-layer bias for this matrix
    const float* featA, const float* featB, const float* featC, const float* featD,
    float* acc0, float* accSV, float* accVS) {
  const int half = lane >> 4;
  const int mrow = lane & 15;
  // Build bf16 A fragments for both edge tiles, both k-chunks (this matrix).
  v16bf a0[2], a1[2];
  #pragma unroll
  for (int et = 0; et < 2; ++et)
    #pragma unroll
    for (int j = 0; j < 16; ++j) {
      int kl = half * 16 + j;
      a0[et][j] = (__bf16)hid[(kl)      * 32 + et * 16 + mrow];
      a1[et][j] = (__bf16)hid[(32 + kl) * 32 + et * 16 + mrow];
    }

  for (int g = wave; g < 144; g += 8) {
    int h = g / 36;
    int r = g % 36;
    int sec, u0, col0;
    if (r < 16)      { sec = 0; u0 = r * 2;        col0 = r * 16; }
    else if (r < 24) { sec = 1; u0 = (r - 16) * 2; col0 = 256 + (r - 16) * 16; }
    else if (r < 32) { sec = 2; u0 = (r - 24) * 4; col0 = 384 + (r - 24) * 16; }
    else             { sec = 3; u0 = (r - 32) * 4; col0 = 512 + (r - 32) * 16; }
    int cgl = (h * 576 + col0) >> 4;

    // One B fetch (both k-chunks) + bias, reused by both edge tiles.
    union { uint4 q[2]; v16bf v; } b0, b1;
    const __bf16* bp = wbase + (size_t)cgl * 1024 + lane * 16;
    b0.q[0] = ((const uint4*)bp)[0];
    b0.q[1] = ((const uint4*)bp)[1];
    b1.q[0] = ((const uint4*)(bp + 512))[0];
    b1.q[1] = ((const uint4*)(bp + 512))[1];
    float bias = b2[h * 576 + col0 + (lane & 15)];

    #pragma unroll
    for (int et = 0; et < 2; ++et) {
      v8f d = {};
      d = wmma_bf16(a0[et], b0.v, d);
      d = wmma_bf16(a1[et], b1.v, d);
      #pragma unroll
      for (int rr = 0; rr < 8; ++rr) d[rr] += bias;

      if (sec == 0 || sec == 1) {  // out0: 2 u x 8 w
        int u = u0 + ((lane >> 3) & 1);
        int w = lane & 7;
        const float* feat =
            (sec == 0) ? &featA[et * 512 + u * 16] : &featB[et * 256 + u * 16];
        float* acc = &acc0[et * 512 + h * 8 + w];
        #pragma unroll
        for (int rr = 0; rr < 8; ++rr)
          atomicAdd(&acc[(m0b + rr) * 32], d[rr] * feat[m0b + rr]);
      } else if (sec == 2) {  // sv: 4 u x 4 w
        int u = u0 + ((lane >> 2) & 3);
        int w = lane & 3;
        const float* feat = &featC[et * 512 + u * 16];
        float* acc = &accSV[et * 256 + h * 4 + w];
        #pragma unroll
        for (int rr = 0; rr < 8; ++rr)
          atomicAdd(&acc[(m0b + rr) * 16], d[rr] * feat[m0b + rr]);
      } else {  // vs: 4 u x 4 w, 3 i-components
        int u = u0 + ((lane >> 2) & 3);
        int w = lane & 3;
        #pragma unroll
        for (int ii = 0; ii < 3; ++ii) {
          const float* feat = &featD[et * 768 + ii * 256 + u * 16];
          float* acc = &accVS[et * 768 + (h * 4 + w) * 3 + ii];
          #pragma unroll
          for (int rr = 0; rr < 8; ++rr)
            atomicAdd(&acc[(m0b + rr) * 48], d[rr] * feat[m0b + rr]);
        }
      }
    }
  }
}

// ---------------------------------------------------------------------------
// Edge kernel: fused radial-MLP (WMMA bf16) + tensor product + logits.
// 1 block = 32 edges (two 16-row WMMA M-tiles), 256 threads = 8 waves.
// ---------------------------------------------------------------------------
__global__ __launch_bounds__(256) void edge_kernel(
    const float* __restrict__ nf, const int* __restrict__ ei,
    const float* __restrict__ esh, const float* __restrict__ emb,
    const float* __restrict__ Wk1, const float* __restrict__ bk1,
    const float* __restrict__ bk2,
    const float* __restrict__ Wv1, const float* __restrict__ bv1,
    const float* __restrict__ bv2,
    const float* __restrict__ Wd0, const float* __restrict__ Wd1,
    const __bf16* __restrict__ w2bf,
    const float* __restrict__ q0g, const float* __restrict__ q1g,
    float* __restrict__ logits, float* __restrict__ v0o, float* __restrict__ v1o,
    int E) {
  __shared__ float sFeatA[2 * 512];   // (ss*sh0)[et][u][el]
  __shared__ float sFeatB[2 * 256];   // vv_dot[et][u][el]
  __shared__ float sFeatC[2 * 512];   // ss[et][u][el]
  __shared__ float sFeatD[2 * 768];   // (vv*sh0)[et][i][u][el] (raw vv first)
  __shared__ float sSh0[32];
  __shared__ float sSh1[32 * 3];
  __shared__ float sEmb[32 * 16];     // [e][b]
  __shared__ float sHid[2 * 64 * 32]; // [mat][k][e]
  __shared__ float sAcc0[2 * 2 * 512];   // [mat][et][el][h*8+w]
  __shared__ float sAccSV[2 * 2 * 256];  // [mat][et][el][h*4+w]
  __shared__ float sAccVS[2 * 2 * 768];  // [mat][et][el][(h*4+w)*3+i]
  __shared__ int   sDst[32];

  const int tid = threadIdx.x;
  const int e0  = blockIdx.x * 32;

  for (int i = tid; i < 2 * 2 * 512, i < 2048; i += 256) sAcc0[i]  = 0.f;
  for (int i = tid; i < 1024; i += 256) sAccSV[i] = 0.f;
  for (int i = tid; i < 3072; i += 256) sAccVS[i] = 0.f;

  if (tid < 32) {
    int e = min(e0 + tid, E - 1);
    sDst[tid]         = ei[E + e];
    sSh0[tid]         = esh[e * 4 + 0];
    sSh1[tid * 3 + 0] = esh[e * 4 + 1];
    sSh1[tid * 3 + 1] = esh[e * 4 + 2];
    sSh1[tid * 3 + 2] = esh[e * 4 + 3];
  }
  for (int i = tid; i < 32 * 16; i += 256) {
    int e = i >> 4, b = i & 15;
    int eg = min(e0 + e, E - 1);
    sEmb[e * 16 + b] = emb[eg * 16 + b];
  }
  for (int i = tid; i < 32 * 80; i += 256) {  // gather src features
    int e = i / 80, c = i % 80;
    int et = e >> 4, el = e & 15;
    int eg = min(e0 + e, E - 1);
    int sIdx = ei[eg];
    float val = nf[(size_t)sIdx * 80 + c];
    if (c < 32) sFeatC[et * 512 + c * 16 + el] = val;
    else {
      int u = (c - 32) / 3, ii = (c - 32) % 3;
      sFeatD[et * 768 + ii * 256 + u * 16 + el] = val;
    }
  }
  __syncthreads();

  for (int t = tid; t < 512; t += 256) {  // vv_dot (raw vv still in sFeatD)
    int u = (t >> 5) & 15, e = t & 31;
    int et = e >> 4, el = e & 15;
    float dsum = 0.f;
    #pragma unroll
    for (int i = 0; i < 3; ++i)
      dsum += sFeatD[et * 768 + i * 256 + u * 16 + el] * sSh1[e * 3 + i];
    sFeatB[et * 256 + u * 16 + el] = dsum * INV_SQRT3;
  }
  __syncthreads();
  for (int i = tid; i < 2 * 768; i += 256) {  // featD = vv*sh0
    int et = i / 768, el = i & 15;
    sFeatD[i] *= sSh0[et * 16 + el];
  }
  for (int i = tid; i < 2 * 512; i += 256) {  // featA = ss*sh0
    int et = i / 512, el = i & 15;
    sFeatA[i] = sFeatC[i] * sSh0[et * 16 + el];
  }

  // Layer-1 MLP: hid[mat][j][e] = silu(emb @ W1 + b1)
  for (int t = tid; t < 2 * 64 * 32; t += 256) {
    int mat = t >> 11, j = (t >> 5) & 63, e = t & 31;
    const float* W1 = mat ? Wv1 : Wk1;
    const float* b1 = mat ? bv1 : bk1;
    float acc = b1[j];
    #pragma unroll
    for (int b = 0; b < 16; ++b) acc += sEmb[e * 16 + b] * W1[b * 64 + j];
    sHid[mat * 2048 + j * 32 + e] = siluf(acc);
  }
  __syncthreads();

  const int lane = tid & 31;
  const int wave = tid >> 5;
  const int m0b  = (lane >> 4) * 8;
  // K matrix (mat=0)
  tp_groups(wave, lane, m0b, sHid, w2bf, bk2, sFeatA, sFeatB, sFeatC, sFeatD,
            sAcc0, sAccSV, sAccVS);
  // V matrix (mat=1)
  tp_groups(wave, lane, m0b, sHid + 2048, w2bf + 147456, bv2, sFeatA, sFeatB, sFeatC,
            sFeatD, sAcc0 + 1024, sAccSV + 512, sAccVS + 1536);
  __syncthreads();

  // Emit v0, v1 (mat=1)
  for (int t = tid; t < 32 * 32; t += 256) {
    int e = t >> 5, hw = t & 31, eg = e0 + e;
    int et = e >> 4, el = e & 15;
    if (eg < E)
      v0o[(size_t)eg * 32 + hw] = sAcc0[1024 + et * 512 + el * 32 + hw] * INV_FAN;
  }
  for (int t = tid; t < 32 * 48; t += 256) {
    int e = t / 48, rest = t % 48, hw = rest / 3, ii = rest % 3, eg = e0 + e;
    int et = e >> 4, el = e & 15;
    if (eg < E)
      v1o[(size_t)eg * 48 + rest] =
          (sAccSV[512 + et * 256 + el * 16 + hw] * sSh1[e * 3 + ii] +
           sAccVS[1536 + et * 768 + el * 48 + rest]) * INV_FAN;
  }
  // Logits from k (mat=0) and gathered q[dst]
  if (tid < 128) {
    int e = tid >> 2, h = tid & 3, eg = e0 + e;
    int et = e >> 4, el = e & 15;
    if (eg < E) {
      int dn = sDst[e];
      float k0[8];
      #pragma unroll
      for (int w = 0; w < 8; ++w)
        k0[w] = sAcc0[et * 512 + el * 32 + h * 8 + w] * INV_FAN;
      float l0 = 0.f;
      #pragma unroll
      for (int u = 0; u < 8; ++u) {
        float qv = q0g[(size_t)dn * 32 + h * 8 + u];
        #pragma unroll
        for (int v = 0; v < 8; ++v) l0 += qv * Wd0[u * 8 + v] * k0[v];
      }
      float k1v[4][3];
      #pragma unroll
      for (int v = 0; v < 4; ++v)
        #pragma unroll
        for (int ii = 0; ii < 3; ++ii)
          k1v[v][ii] = (sAccSV[et * 256 + el * 16 + h * 4 + v] * sSh1[e * 3 + ii] +
                        sAccVS[et * 768 + el * 48 + (h * 4 + v) * 3 + ii]) * INV_FAN;
      float l1 = 0.f;
      #pragma unroll
      for (int u = 0; u < 4; ++u)
        #pragma unroll
        for (int ii = 0; ii < 3; ++ii) {
          float qv = q1g[(size_t)dn * 48 + (h * 4 + u) * 3 + ii];
          #pragma unroll
          for (int v = 0; v < 4; ++v) l1 += qv * Wd1[u * 4 + v] * k1v[v][ii];
        }
      float lg = (l0 + l1 * INV_SQRT3) * INV_FAND * INV_SQRT_HD;
      lg = fminf(fmaxf(lg, -10.f), 10.f);
      logits[(size_t)eg * 4 + h] = lg;
    }
  }
}

// ---------------------------------------------------------------------------
// Segment softmax + aggregation
// ---------------------------------------------------------------------------
static __device__ __forceinline__ void atomicMaxF(float* addr, float v) {
  if (v >= 0.f) atomicMax((int*)addr, __float_as_int(v));
  else          atomicMin((unsigned int*)addr, __float_as_uint(v));
}

__global__ __launch_bounds__(256) void smax_kernel(const float* __restrict__ logits,
                                                   const int* __restrict__ ei,
                                                   float* __restrict__ segmax, int E) {
  int idx = blockIdx.x * 256 + threadIdx.x;
  if (idx >= E * 4) return;
  int e = idx >> 2, h = idx & 3;
  atomicMaxF(&segmax[ei[E + e] * 4 + h], logits[idx]);
}

__global__ __launch_bounds__(256) void sexp_kernel(const float* __restrict__ logits,
                                                   const int* __restrict__ ei,
                                                   const float* __restrict__ segmax,
                                                   float* __restrict__ exb,
                                                   float* __restrict__ denom, int E) {
  int idx = blockIdx.x * 256 + threadIdx.x;
  if (idx >= E * 4) return;
  int e = idx >> 2, h = idx & 3, d = ei[E + e];
  float ex = __expf(logits[idx] - segmax[d * 4 + h]);
  exb[idx] = ex;
  atomicAdd(&denom[d * 4 + h], ex);
}

__global__ __launch_bounds__(256) void agg_kernel(const float* __restrict__ exb,
                                                  const float* __restrict__ denom,
                                                  const int* __restrict__ ei,
                                                  const float* __restrict__ v0o,
                                                  const float* __restrict__ v1o,
                                                  float* __restrict__ agg0,
                                                  float* __restrict__ agg1, int E) {
  int idx = blockIdx.x * 256 + threadIdx.x;
  if (idx >= E * 4) return;
  int e = idx >> 2, h = idx & 3, d = ei[E + e];
  float alpha = exb[idx] / (denom[d * 4 + h] + 1e-12f);
  #pragma unroll
  for (int w = 0; w < 8; ++w)
    atomicAdd(&agg0[(size_t)d * 32 + h * 8 + w], alpha * v0o[(size_t)e * 32 + h * 8 + w]);
  #pragma unroll
  for (int t = 0; t < 12; ++t)
    atomicAdd(&agg1[(size_t)d * 48 + h * 12 + t], alpha * v1o[(size_t)e * 48 + h * 12 + t]);
}

// ---------------------------------------------------------------------------
// Node output: residual projections + gated MLP. One 64-thread block per node.
// ---------------------------------------------------------------------------
__global__ __launch_bounds__(64) void node_kernel(
    const float* __restrict__ nf, const float* __restrict__ agg0,
    const float* __restrict__ agg1, const float* __restrict__ Wo0,
    const float* __restrict__ Wo1, const float* __restrict__ Wf10,
    const float* __restrict__ Wf11, const float* __restrict__ Wf20,
    const float* __restrict__ Wf21, float* __restrict__ out, int N) {
  __shared__ float x0[32], x1[48], a0[64], a1[48];
  int n = blockIdx.x, t = threadIdx.x;
  const float* row = nf + (size_t)n * 80;
  if (t < 32) {
    float acc = 0.f;
    #pragma unroll
    for (int j = 0; j < 32; ++j) acc += agg0[(size_t)n * 32 + j] * Wo0[j * 32 + t];
    x0[t] = row[t] + acc * INV_SQRT32;
  }
  if (t < 48) {
    int u = t / 3, ii = t % 3;
    float acc = 0.f;
    #pragma unroll
    for (int j = 0; j < 16; ++j) acc += agg1[(size_t)n * 48 + j * 3 + ii] * Wo1[j * 16 + u];
    x1[t] = row[32 + t] + acc * 0.25f;
  }
  __syncthreads();
  {
    float acc = 0.f;
    #pragma unroll
    for (int u = 0; u < 32; ++u) acc += x0[u] * Wf10[u * 64 + t];
    acc *= INV_SQRT32;
    a0[t] = siluf(acc);
  }
  if (t < 48) {
    int u = t / 3, ii = t % 3;
    float acc = 0.f;
    #pragma unroll
    for (int p = 0; p < 16; ++p) acc += x1[p * 3 + ii] * Wf11[p * 16 + u];
    a1[t] = acc * 0.25f;
  }
  __syncthreads();
  if (t < 16) {
    float nx = a1[t * 3], ny = a1[t * 3 + 1], nz = a1[t * 3 + 2];
    float nm = sqrtf(nx * nx + ny * ny + nz * nz);
    float g = (nm < 1e-8f) ? 0.f : siluf(nm) / fmaxf(nm, 1e-8f);
    a1[t * 3] *= g; a1[t * 3 + 1] *= g; a1[t * 3 + 2] *= g;
  }
  __syncthreads();
  if (t < 32) {
    float acc = 0.f;
    #pragma unroll
    for (int c = 0; c < 64; ++c) acc += a0[c] * Wf20[c * 32 + t];
    out[(size_t)n * 80 + t] = x0[t] + acc * 0.125f;
  }
  if (t < 48) {
    int u = t / 3, ii = t % 3;
    float acc = 0.f;
    #pragma unroll
    for (int p = 0; p < 16; ++p) acc += a1[p * 3 + ii] * Wf21[p * 16 + u];
    out[(size_t)n * 80 + 32 + t] = x1[t] + acc * 0.25f;
  }
}

// ---------------------------------------------------------------------------
extern "C" void kernel_launch(void* const* d_in, const int* in_sizes, int n_in,
                              void* d_out, int out_size, void* d_ws, size_t ws_size,
                              hipStream_t stream) {
  const float* nf   = (const float*)d_in[0];
  const int*   ei   = (const int*)d_in[1];
  const float* esh  = (const float*)d_in[2];
  const float* emb  = (const float*)d_in[3];
  const float* Wq0  = (const float*)d_in[4];
  const float* Wq1  = (const float*)d_in[5];
  const float* Wk1  = (const float*)d_in[6];
  const float* bk1  = (const float*)d_in[7];
  const float* Wk2  = (const float*)d_in[8];
  const float* bk2  = (const float*)d_in[9];
  const float* Wv1  = (const float*)d_in[10];
  const float* bv1  = (const float*)d_in[11];
  const float* Wv2  = (const float*)d_in[12];
  const float* bv2  = (const float*)d_in[13];
  const float* Wd0  = (const float*)d_in[14];
  const float* Wd1  = (const float*)d_in[15];
  const float* Wo0  = (const float*)d_in[16];
  const float* Wo1  = (const float*)d_in[17];
  const float* Wf10 = (const float*)d_in[18];
  const float* Wf11 = (const float*)d_in[19];
  const float* Wf20 = (const float*)d_in[20];
  const float* Wf21 = (const float*)d_in[21];

  int N = in_sizes[0] / 80;
  int E = in_sizes[1] / 2;

  char* ws = (char*)d_ws;
  size_t off = 0;
  __bf16* w2bf = (__bf16*)(ws + off); off += (size_t)2 * 147456 * 2;
  off = (off + 255) & ~(size_t)255;
  float* q0g    = (float*)(ws + off); off += (size_t)N * 32 * 4;
  float* q1g    = (float*)(ws + off); off += (size_t)N * 48 * 4;
  float* logits = (float*)(ws + off); off += (size_t)E * 4 * 4;
  float* exb    = (float*)(ws + off); off += (size_t)E * 4 * 4;
  float* v0o    = (float*)(ws + off); off += (size_t)E * 32 * 4;
  float* v1o    = (float*)(ws + off); off += (size_t)E * 48 * 4;
  float* segmax = (float*)(ws + off); off += (size_t)N * 4 * 4;
  float* denom  = (float*)(ws + off); off += (size_t)N * 4 * 4;
  float* agg0   = (float*)(ws + off); off += (size_t)N * 32 * 4;
  float* agg1   = (float*)(ws + off); off += (size_t)N * 48 * 4;
  (void)ws_size; (void)n_in; (void)out_size;

  prep_kernel<<<(2 * 147456 + 255) / 256, 256, 0, stream>>>(Wk2, Wv2, w2bf);
  init_kernel<<<(N * 48 + 255) / 256, 256, 0, stream>>>(segmax, denom, agg0, agg1, N);
  q_kernel<<<(N * 80 + 255) / 256, 256, 0, stream>>>(nf, Wq0, Wq1, q0g, q1g, N);

  int EB = (E + 31) / 32;
  edge_kernel<<<EB, 256, 0, stream>>>(nf, ei, esh, emb, Wk1, bk1, bk2, Wv1, bv1, bv2,
                                      Wd0, Wd1, w2bf, q0g, q1g, logits, v0o, v1o, E);

  smax_kernel<<<(E * 4 + 255) / 256, 256, 0, stream>>>(logits, ei, segmax, E);
  sexp_kernel<<<(E * 4 + 255) / 256, 256, 0, stream>>>(logits, ei, segmax, exb, denom, E);
  agg_kernel<<<(E * 4 + 255) / 256, 256, 0, stream>>>(exb, denom, ei, v0o, v1o, agg0, agg1, E);

  node_kernel<<<N, 64, 0, stream>>>(nf, agg0, agg1, Wo0, Wo1, Wf10, Wf11, Wf20, Wf21,
                                    (float*)d_out, N);
}
